// ColbertLoss_32547262169681
// MI455X (gfx1250) — compile-verified, matrix operand order checked by reading, not run on hardware
//
#include <hip/hip_runtime.h>

// ---------------------------------------------------------------------------
// ColBERT loss on gfx1250 (MI455X):
//   scores[q][p] = sum_n qmask[q][n] * max_m( dmask[p][m] ? <Q[q,n,:],D[p,m,:]> : -1e30 ) / T
//   loss = -mean_q( scores[q][q] - logsumexp_p scores[q][p] )
// GEMM via bf16x3 truncation split (hi*hi + hi*lo + lo*hi) on
// v_wmma_f32_16x16x32_bf16, three independent accumulator chains per tile,
// branchless doc-mask fold, packed-dword staging conversion.
// ---------------------------------------------------------------------------

typedef __bf16 v16bf __attribute__((ext_vector_type(16)));
typedef __bf16 v8bf  __attribute__((ext_vector_type(8)));
typedef float  v8f   __attribute__((ext_vector_type(8)));
typedef float  v4f   __attribute__((ext_vector_type(4)));
typedef unsigned uv2 __attribute__((ext_vector_type(2)));

#define BQ 64
#define SQ 64
#define DK 128
#define BD 64
#define SD 512
#define CHUNK 128               // doc tokens staged per LDS chunk
#define NCHUNK (SD / CHUNK)     // 4
#define STR 136                 // padded row stride in bf16 elems (272B -> conflict-free b128)
#define NEGINF (-1.0e30f)
#define INV_TEMP 50.0f          // 1 / 0.02

// Truncation hi/lo split of 4 fp32 into two packed-bf16 dword pairs.
// hi = x & 0xFFFF0000 (top 8 mantissa bits); rem = x - hi is EXACT in fp32
// (<=16 mantissa bits), lo = trunc-bf16(rem). hi*hi + hi*lo + lo*hi then
// reconstructs x*y to ~2^-16 relative, same as an RNE split but far cheaper.
__device__ __forceinline__ void split4(v4f x, uv2* hiw, uv2* low) {
  unsigned u0 = __float_as_uint(x.x), u1 = __float_as_uint(x.y);
  unsigned u2 = __float_as_uint(x.z), u3 = __float_as_uint(x.w);
  (*hiw).x = (u0 >> 16) | (u1 & 0xFFFF0000u);
  (*hiw).y = (u2 >> 16) | (u3 & 0xFFFF0000u);
  float r0 = x.x - __uint_as_float(u0 & 0xFFFF0000u);
  float r1 = x.y - __uint_as_float(u1 & 0xFFFF0000u);
  float r2 = x.z - __uint_as_float(u2 & 0xFFFF0000u);
  float r3 = x.w - __uint_as_float(u3 & 0xFFFF0000u);
  (*low).x = (__float_as_uint(r0) >> 16) | (__float_as_uint(r1) & 0xFFFF0000u);
  (*low).y = (__float_as_uint(r2) >> 16) | (__float_as_uint(r3) & 0xFFFF0000u);
}

// Load one 16x32 bf16 fragment for this lane from padded LDS:
// lane l holds token row (l%16); K halves at k0 and k0+16.
__device__ __forceinline__ v16bf load_frag(const unsigned short* base, int row, int k0) {
  const v8bf* p0 = (const v8bf*)(base + row * STR + k0);
  const v8bf* p1 = (const v8bf*)(base + row * STR + k0 + 16);
  v8bf a = *p0;
  v8bf b = *p1;
  return __builtin_shufflevector(a, b, 0, 1, 2, 3, 4, 5, 6, 7,
                                       8, 9, 10, 11, 12, 13, 14, 15);
}

__global__ __launch_bounds__(256)
void colbert_scores_kernel(const float* __restrict__ qe,
                           const float* __restrict__ de,
                           const unsigned char* __restrict__ qmask,
                           const unsigned char* __restrict__ dmask,
                           float* __restrict__ scores) {
  __shared__ __attribute__((aligned(16))) unsigned short qh[SQ * STR];
  __shared__ __attribute__((aligned(16))) unsigned short ql[SQ * STR];
  __shared__ __attribute__((aligned(16))) unsigned short dh[CHUNK * STR];
  __shared__ __attribute__((aligned(16))) unsigned short dl[CHUNK * STR];
  __shared__ float wmaxbuf[8][16];
  __shared__ float tokbuf[SQ];

  const int p = blockIdx.x;          // doc batch index
  const int q = blockIdx.y;          // query batch index
  const int tid = threadIdx.x;
  const int wave  = tid >> 5;        // 0..7
  const int lane  = tid & 31;
  const int lrow  = lane & 15;       // token row within a tile
  const int lhalf = lane >> 4;       // K-half selector per WMMA layout
  const int mtile = wave & 3;        // this wave's M (query) tile
  const int npar  = wave >> 2;       // N-tile parity

  // ---- stage query block: fp32 -> packed bf16 hi/lo into LDS ----
  const float* qbase = qe + (size_t)q * SQ * DK;
  #pragma unroll
  for (int s = 0; s < (SQ * DK) / (256 * 4); ++s) {   // 8 iters
    int idx4 = tid + s * 256;
    int r = idx4 >> 5;                 // 32 vec4 per row
    int col = (idx4 & 31) * 4;
    v4f x = *(const v4f*)(qbase + r * DK + col);
    uv2 hw, lw;
    split4(x, &hw, &lw);
    *(uv2*)(qh + r * STR + col) = hw;
    *(uv2*)(ql + r * STR + col) = lw;
  }
  __syncthreads();

  // ---- load A (query) fragments once; reused for all 128 output tiles ----
  v16bf ahi[4], alo[4];
  {
    int row = mtile * 16 + lrow;
    #pragma unroll
    for (int k = 0; k < 4; ++k) {
      int k0 = k * 32 + lhalf * 8;
      ahi[k] = load_frag(qh, row, k0);
      alo[k] = load_frag(ql, row, k0);
    }
  }

  float runmax[8];
  #pragma unroll
  for (int j = 0; j < 8; ++j) runmax[j] = NEGINF;

  const float* dbase = de + (size_t)p * SD * DK;
  const unsigned char* dmbase = dmask + (size_t)p * SD + npar * 16 + lrow;
  for (int c = 0; c < NCHUNK; ++c) {
    // this lane's 4 doc-mask column biases for this chunk (no dynamic indexing)
    const unsigned char* dmc = dmbase + c * CHUNK;
    float bias0 = dmc[0]  ? 0.0f : NEGINF;
    float bias1 = dmc[32] ? 0.0f : NEGINF;
    float bias2 = dmc[64] ? 0.0f : NEGINF;
    float bias3 = dmc[96] ? 0.0f : NEGINF;

    __syncthreads();  // dh/dl reuse guard
    #pragma unroll
    for (int s = 0; s < (CHUNK * DK) / (256 * 4); ++s) {  // 16 iters
      int idx4 = tid + s * 256;
      int r = idx4 >> 5;
      int col = (idx4 & 31) * 4;
      v4f x = *(const v4f*)(dbase + (size_t)(c * CHUNK + r) * DK + col);
      uv2 hw, lw;
      split4(x, &hw, &lw);
      *(uv2*)(dh + r * STR + col) = hw;
      *(uv2*)(dl + r * STR + col) = lw;
    }
    __syncthreads();

    #pragma unroll
    for (int i = 0; i < 4; ++i) {
      int ntile = npar + 2 * i;                 // 0..7 within chunk
      int brow = ntile * 16 + lrow;
      // three independent WMMA chains (depth 4 each) for ILP
      v8f acc0 = {0.f, 0.f, 0.f, 0.f, 0.f, 0.f, 0.f, 0.f};
      v8f acc1 = acc0;
      v8f acc2 = acc0;
      #pragma unroll
      for (int k = 0; k < 4; ++k) {
        int k0 = k * 32 + lhalf * 8;
        v16bf bhi = load_frag(dh, brow, k0);
        v16bf blo = load_frag(dl, brow, k0);
        acc0 = __builtin_amdgcn_wmma_f32_16x16x32_bf16(false, ahi[k], false, bhi,
                                                       (short)0, acc0, false, false);
        acc1 = __builtin_amdgcn_wmma_f32_16x16x32_bf16(false, ahi[k], false, blo,
                                                       (short)0, acc1, false, false);
        acc2 = __builtin_amdgcn_wmma_f32_16x16x32_bf16(false, alo[k], false, bhi,
                                                       (short)0, acc2, false, false);
      }
      // branchless doc-token max fold: masked columns get -1e30 bias
      float bias = (i == 0) ? bias0 : (i == 1) ? bias1 : (i == 2) ? bias2 : bias3;
      #pragma unroll
      for (int j = 0; j < 8; ++j) {
        float cand = (acc0[j] + acc1[j] + acc2[j]) + bias;
        runmax[j] = fmaxf(runmax[j], cand);
      }
    }
  }

  // ---- cross-lane max over the 16 columns each lane-group holds ----
  #pragma unroll
  for (int j = 0; j < 8; ++j) {
    float v = runmax[j];
    #pragma unroll
    for (int m = 8; m >= 1; m >>= 1) v = fmaxf(v, __shfl_xor(v, m, 32));
    runmax[j] = v;
  }
  // C layout: VGPR j -> row j (lanes 0-15) / row j+8 (lanes 16-31)
  if (lrow == 0) {
    #pragma unroll
    for (int j = 0; j < 8; ++j) wmaxbuf[wave][lhalf * 8 + j] = runmax[j];
  }
  __syncthreads();

  if (tid < SQ) {
    int mt = tid >> 4, r = tid & 15;
    float v = fmaxf(wmaxbuf[mt][r], wmaxbuf[mt + 4][r]);  // waves w and w+4 share M tile
    tokbuf[tid] = qmask[(size_t)q * SQ + tid] ? v : 0.f;
  }
  __syncthreads();
  if (tid == 0) {
    float s = 0.f;
    for (int i = 0; i < SQ; ++i) s += tokbuf[i];
    scores[q * BD + p] = s * INV_TEMP;
  }
}

__global__ __launch_bounds__(64)
void colbert_loss_kernel(const float* __restrict__ scores, float* __restrict__ out) {
  __shared__ float buf[BQ];
  int qi = threadIdx.x;
  const float* row = scores + qi * BD;
  float m = NEGINF;
  for (int pj = 0; pj < BD; ++pj) m = fmaxf(m, row[pj]);
  float s = 0.f;
  for (int pj = 0; pj < BD; ++pj) s += __expf(row[pj] - m);
  float lse = m + __logf(s);
  buf[qi] = row[qi] - lse;          // diagonal log-prob
  __syncthreads();
  if (qi == 0) {
    float t = 0.f;
    for (int i = 0; i < BQ; ++i) t += buf[i];
    out[0] = -t / (float)BQ;
  }
}

extern "C" void kernel_launch(void* const* d_in, const int* in_sizes, int n_in,
                              void* d_out, int out_size, void* d_ws, size_t ws_size,
                              hipStream_t stream) {
  const float* qe = (const float*)d_in[0];                  // (64,64,128) f32
  const float* de = (const float*)d_in[1];                  // (64,512,128) f32
  const unsigned char* qm = (const unsigned char*)d_in[2];  // (64,64) bool
  const unsigned char* dm = (const unsigned char*)d_in[3];  // (64,512) bool
  float* scores = (float*)d_ws;                             // 64*64 f32 scratch

  dim3 grid(BD, BQ);
  colbert_scores_kernel<<<grid, 256, 0, stream>>>(qe, de, qm, dm, scores);
  colbert_loss_kernel<<<1, 64, 0, stream>>>(scores, (float*)d_out);
}
